// KimiDeltaAttentionFLA_27788438405524
// MI455X (gfx1250) — matrix operand order repaired and checked by reading, not compiled, not run on hardware
//
#include <hip/hip_runtime.h>

#define BB    2
#define TSEQ  2048
#define HIDD  2048
#define HH    16
#define KK    128
#define DD    (HH * KK)      // 2048
#define KCV   4
#define NTOK  (BB * TSEQ)    // 4096
#define EPSV  1e-5f
#define QSCALE 0.08838834764831845f  // 128^-0.5

typedef __attribute__((ext_vector_type(16))) __bf16 v16bf;
typedef __attribute__((ext_vector_type(8)))  float  v8f;
typedef unsigned int u32x4 __attribute__((ext_vector_type(4)));
typedef int          i32x4 __attribute__((ext_vector_type(4)));
typedef int          i32x8 __attribute__((ext_vector_type(8)));

#if defined(__has_builtin)
#if __has_builtin(__builtin_amdgcn_tensor_load_to_lds) && \
    __has_builtin(__builtin_amdgcn_s_wait_tensorcnt)
#define USE_TDM 1
#endif
#endif

// ---------------------------------------------------------------------------
// f32 -> bf16 pack convert (float4 in, 4x bf16 out per thread)
// ---------------------------------------------------------------------------
__global__ __launch_bounds__(256)
void cvt_bf16_kernel(const float* __restrict__ in, __bf16* __restrict__ out, int n4)
{
    int i = blockIdx.x * 256 + threadIdx.x;
    if (i >= n4) return;
    float4 v = ((const float4*)in)[i];
    union { __bf16 h[4]; uint2 u; } pk;
    pk.h[0] = (__bf16)v.x; pk.h[1] = (__bf16)v.y;
    pk.h[2] = (__bf16)v.z; pk.h[3] = (__bf16)v.w;
    ((uint2*)out)[i] = pk.u;
}

// ---------------------------------------------------------------------------
// f32 [R][C] -> bf16 transposed [C][R], 32x32 LDS tiles.
// ---------------------------------------------------------------------------
__global__ __launch_bounds__(256)
void cvt_transpose_bf16_kernel(const float* __restrict__ in, __bf16* __restrict__ out,
                               int R, int Ccols)
{
    __shared__ __bf16 tile[32][33];
    int bx = blockIdx.x * 32;            // col base in input
    int by = blockIdx.y * 32;            // row base in input
    int tx = threadIdx.x & 31;
    int ty = (threadIdx.x >> 5) * 4;     // 8 waves x 4 rows
    #pragma unroll
    for (int i = 0; i < 4; ++i) {
        int r = by + ty + i, c = bx + tx;
        float v = (r < R && c < Ccols) ? in[(size_t)r * Ccols + c] : 0.f;
        tile[ty + i][tx] = (__bf16)v;
    }
    __syncthreads();
    #pragma unroll
    for (int i = 0; i < 4; ++i) {
        int c = bx + ty + i;             // output row = input col
        int r = by + tx;                 // output col = input row
        if (c < Ccols && r < R)
            out[(size_t)c * R + r] = tile[tx][ty + i];
    }
}

// ---------------------------------------------------------------------------
// WMMA GEMM: C[M,N] = A[M,Kd] * BT[N,Kd]^T; bf16 inputs (B pre-transposed),
// f32 out. Block: 256 threads (8 waves), tile 128x128x64, double-buffered LDS.
// Staging via Tensor Data Mover (TENSORcnt) when available; TDM pad fields
// reproduce the 72-element (144B) padded row stride the fragments expect.
// ---------------------------------------------------------------------------
#define GBM 128
#define GBN 128
#define GBK 64
#define GBKP 72   // padded k-stride (144 B, multiple of 16 B)

#ifdef USE_TDM
__device__ inline void tdm_load_tile(const __bf16* gsrc, unsigned lds_off,
                                     int rows_total, int kd, int row0, int k0)
{
    unsigned long long ga =
        (unsigned long long)(size_t)(gsrc + (size_t)row0 * kd + k0);
    u32x4 g0;
    g0[0] = 1u;                                   // count=1 (valid user D#)
    g0[1] = lds_off;                              // LDS byte address
    g0[2] = (unsigned)(ga & 0xffffffffu);         // global_addr[31:0]
    g0[3] = (unsigned)((ga >> 32) & 0x1ffffffu)   // global_addr[56:32]
          | 0x80000000u;                          // type=2 ("image")
    unsigned dim0 = (unsigned)kd;                 // tensor dim0 (elements)
    unsigned dim1 = (unsigned)rows_total;         // tensor dim1 (rows, OOB clamp)
    unsigned long long st0 = (unsigned long long)kd;
    i32x8 g1;
    g1[0] = (int)((1u << 16)        // data_size = 2 bytes (bf16)
                | (1u << 20)        // pad_enable
                | (4u << 22)        // pad_interval: 32 DWORDs = 128 B (one row)
                | (3u << 25));      // pad_amount: 4 DWORDs = 16 B
    g1[1] = (int)((dim0 & 0xffffu) << 16);
    g1[2] = (int)((dim0 >> 16) | ((dim1 & 0xffffu) << 16));
    g1[3] = (int)((dim1 >> 16) | ((unsigned)GBK << 16));   // tile_dim0 = 64
    g1[4] = (int)((unsigned)GBM);                          // tile_dim1=128, dim2=0
    g1[5] = (int)(st0 & 0xffffffffu);                      // dim0 stride
    g1[6] = (int)((st0 >> 32) & 0xffffu);
    g1[7] = 0;
    i32x4 gz = {0, 0, 0, 0};
#if __clang_major__ >= 23
    i32x8 gz8 = {0, 0, 0, 0, 0, 0, 0, 0};
    __builtin_amdgcn_tensor_load_to_lds(g0, g1, gz, gz, gz8, 0);
#else
    __builtin_amdgcn_tensor_load_to_lds(g0, g1, gz, gz, 0);
#endif
}
#endif

__global__ __launch_bounds__(256)
void gemm_wmma_bf16(const __bf16* __restrict__ A, const __bf16* __restrict__ BT,
                    float* __restrict__ C, int M, int N, int Kd)
{
    __shared__ __bf16 As[2][GBM][GBKP];   // 2 x 18 KB
    __shared__ __bf16 Bs[2][GBN][GBKP];   // 2 x 18 KB   (Bs[n][k])

    const int tid  = threadIdx.x;
    const int lane = tid & 31;
    const int wv   = tid >> 5;
    const int wm   = wv & 3;          // 4 wave-rows * 32 rows
    const int wn   = wv >> 2;         // 2 wave-cols * 64 cols
    const int half = lane >> 4;
    const int l15  = lane & 15;
    const int m0   = blockIdx.y * GBM;
    const int n0   = blockIdx.x * GBN;
    const int nK   = Kd / GBK;

    v8f acc[2][4] = {};

#ifdef USE_TDM
    // ---- prologue: TDM-stage k-tile 0 into buffer 0 ----
    if (wv == 0) {
        tdm_load_tile(A,  (unsigned)(size_t)&As[0][0][0], M, Kd, m0, 0);
        tdm_load_tile(BT, (unsigned)(size_t)&Bs[0][0][0], N, Kd, n0, 0);
    }
#endif

    for (int kt = 0; kt < nK; ++kt) {
        const int cur = kt & 1;
#ifdef USE_TDM
        if (wv == 0) {
            if (kt + 1 < nK) {
                // issue next stage into the other buffer, then retire current
                tdm_load_tile(A,  (unsigned)(size_t)&As[cur ^ 1][0][0],
                              M, Kd, m0, (kt + 1) * GBK);
                tdm_load_tile(BT, (unsigned)(size_t)&Bs[cur ^ 1][0][0],
                              N, Kd, n0, (kt + 1) * GBK);
                __builtin_amdgcn_s_wait_tensorcnt(2);
            } else {
                __builtin_amdgcn_s_wait_tensorcnt(0);
            }
        }
        __syncthreads();
#else
        // ---- synchronous staging fallback: A and BT are both row-major ----
        int k0 = kt * GBK;
        #pragma unroll
        for (int it = 0; it < 4; ++it) {
            int idx = (tid + it * 256) * 8;
            int r = idx >> 6, c = idx & 63;
            int gm = m0 + r;
            uint4 v = make_uint4(0u, 0u, 0u, 0u);
            if (gm < M)
                v = *(const uint4*)(A + (size_t)gm * Kd + k0 + c);
            *(uint4*)(&As[cur][r][c]) = v;
        }
        #pragma unroll
        for (int it = 0; it < 4; ++it) {
            int idx = (tid + it * 256) * 8;
            int r = idx >> 6, c = idx & 63;
            int gn = n0 + r;
            uint4 v = make_uint4(0u, 0u, 0u, 0u);
            if (gn < N)
                v = *(const uint4*)(BT + (size_t)gn * Kd + k0 + c);
            *(uint4*)(&Bs[cur][r][c]) = v;
        }
        __syncthreads();
#endif

        #pragma unroll
        for (int ks = 0; ks < 2; ++ks) {
            union { v16bf v; uint4 q[2]; } afu[2], bfu[4];
            // A fragment (16x32): elems 0..7 -> k = ks*32+half*8..+7,
            //                     elems 8..15 -> +16. Two b128 loads.
            #pragma unroll
            for (int a = 0; a < 2; ++a) {
                int rA = wm * 32 + a * 16 + l15;
                int kb = ks * 32 + half * 8;
                afu[a].q[0] = *(const uint4*)(&As[cur][rA][kb]);
                afu[a].q[1] = *(const uint4*)(&As[cur][rA][kb + 16]);
            }
            // B fragment (32x16): lane n, elems 0..7 -> k = ks*32+half*8..+7, 8..15 -> +16.
            #pragma unroll
            for (int b = 0; b < 4; ++b) {
                int cB = wn * 64 + b * 16 + l15;
                int kb = ks * 32 + half * 8;
                bfu[b].q[0] = *(const uint4*)(&Bs[cur][cB][kb]);
                bfu[b].q[1] = *(const uint4*)(&Bs[cur][cB][kb + 16]);
            }
            #pragma unroll
            for (int a = 0; a < 2; ++a)
                #pragma unroll
                for (int b = 0; b < 4; ++b)
                    acc[a][b] = __builtin_amdgcn_wmma_f32_16x16x32_bf16(
                        false, afu[a].v, false, bfu[b].v, (short)0, acc[a][b],
                        false, false);
        }
        __syncthreads();
    }

    // Store C: VGPR r holds row half*8+r of each 16x16 tile, col = lane&15.
    #pragma unroll
    for (int a = 0; a < 2; ++a)
        #pragma unroll
        for (int b = 0; b < 4; ++b)
            #pragma unroll
            for (int r = 0; r < 8; ++r) {
                int m = m0 + wm * 32 + a * 16 + half * 8 + r;
                int n = n0 + wn * 64 + b * 16 + l15;
                if (m < M && n < N) C[(size_t)m * N + n] = acc[a][b][r];
            }
}

// ---------------------------------------------------------------------------
// Depthwise causal conv (KC=4) + bias + SiLU.  x,y: [B,T,D]
// ---------------------------------------------------------------------------
__global__ __launch_bounds__(256)
void conv_silu_kernel(const float* __restrict__ x, const float* __restrict__ w,
                      const float* __restrict__ bias, float* __restrict__ y)
{
    int idx = blockIdx.x * 256 + threadIdx.x;
    if (idx >= BB * TSEQ * DD) return;
    int d = idx % DD;
    int t = (idx / DD) % TSEQ;
    float acc = bias[d];
    #pragma unroll
    for (int j = 0; j < KCV; ++j) {
        int dt = j - (KCV - 1);            // -3..0
        float xv = (t + dt >= 0) ? x[idx + dt * DD] : 0.f;
        acc = fmaf(xv, w[d * KCV + j], acc);
    }
    y[idx] = acc / (1.f + __expf(-acc));   // SiLU
}

// ---------------------------------------------------------------------------
// g = -exp(A_log[h]) * softplus(v + dt_bias[d]),  in-place on [B,T,D]
// ---------------------------------------------------------------------------
__global__ __launch_bounds__(256)
void gate_kernel(float* __restrict__ gio, const float* __restrict__ dt_bias,
                 const float* __restrict__ A_log)
{
    int idx = blockIdx.x * 256 + threadIdx.x;
    if (idx >= BB * TSEQ * DD) return;
    int d = idx % DD;
    int h = d / KK;
    float v = gio[idx] + dt_bias[d];
    float sp = (v > 20.f) ? v : log1pf(__expf(v));
    gio[idx] = -__expf(A_log[h]) * sp;
}

__global__ __launch_bounds__(256)
void sigmoid_kernel(float* __restrict__ p, int n)
{
    int i = blockIdx.x * 256 + threadIdx.x;
    if (i < n) p[i] = 1.f / (1.f + __expf(-p[i]));
}

// ---------------------------------------------------------------------------
// L2 normalize rows of length K=128 (wave per row), optional scale.
// ---------------------------------------------------------------------------
__global__ __launch_bounds__(256)
void l2norm_kernel(float* __restrict__ x, float scale)
{
    int row  = blockIdx.x * 8 + (threadIdx.x >> 5);
    int lane = threadIdx.x & 31;
    size_t off = (size_t)row * KK + lane * 4;
    float4 v = *(const float4*)(x + off);
    float s = v.x * v.x + v.y * v.y + v.z * v.z + v.w * v.w;
    #pragma unroll
    for (int o = 16; o > 0; o >>= 1) s += __shfl_xor(s, o, 32);
    float r = rsqrtf(s + 1e-6f) * scale;
    v.x *= r; v.y *= r; v.z *= r; v.w *= r;
    *(float4*)(x + off) = v;
}

// ---------------------------------------------------------------------------
// Gated head-wise RMSNorm: o = o * rsqrt(mean(o^2)+eps) * w * sigmoid(gate)
// ---------------------------------------------------------------------------
__global__ __launch_bounds__(256)
void rmsnorm_gate_kernel(float* __restrict__ o, const float* __restrict__ gate,
                         const float* __restrict__ w)
{
    int row  = blockIdx.x * 8 + (threadIdx.x >> 5);
    int lane = threadIdx.x & 31;
    size_t off = (size_t)row * KK + lane * 4;
    float4 v  = *(const float4*)(o + off);
    float4 gt = *(const float4*)(gate + off);
    float4 wv = *(const float4*)(w + lane * 4);
    float s = v.x * v.x + v.y * v.y + v.z * v.z + v.w * v.w;
    #pragma unroll
    for (int t = 16; t > 0; t >>= 1) s += __shfl_xor(s, t, 32);
    float r = rsqrtf(s * (1.f / KK) + EPSV);
    v.x = v.x * r * wv.x * (1.f / (1.f + __expf(-gt.x)));
    v.y = v.y * r * wv.y * (1.f / (1.f + __expf(-gt.y)));
    v.z = v.z * r * wv.z * (1.f / (1.f + __expf(-gt.z)));
    v.w = v.w * r * wv.w * (1.f / (1.f + __expf(-gt.w)));
    *(float4*)(o + off) = v;
}

// ---------------------------------------------------------------------------
// KDA gated delta-rule scan. Each v-column of S[k,v] is an independent
// recurrence: one thread per (b,h,v-col), state S[:,j] (128 f32) in VGPRs.
// Block = 128 threads = one (b,h); k/q/exp(g) staged in LDS as broadcasts.
// Software-pipelined: t+1's global loads issue before t's 384-FMA compute
// phase so HBM/L2 latency hides under the recurrence arithmetic.
// ---------------------------------------------------------------------------
__global__ __launch_bounds__(128)
void kda_scan_kernel(const float* __restrict__ q, const float* __restrict__ k,
                     const float* __restrict__ v, const float* __restrict__ g,
                     const float* __restrict__ beta, float* __restrict__ o)
{
    int bh = blockIdx.x;              // 0..B*H-1
    int b  = bh / HH, h = bh % HH;
    int j  = threadIdx.x;             // v-column 0..127

    __shared__ float qs[KK], ks[KK], eg[KK];

    float S[KK];
    #pragma unroll
    for (int i = 0; i < KK; ++i) S[i] = 0.f;

    size_t base    = (size_t)b * TSEQ * HH * KK + (size_t)h * KK;
    size_t strideT = (size_t)HH * KK;
    size_t bbase   = (size_t)b * TSEQ * HH + h;

    // preload t = 0
    float qv = q[base + j], kv = k[base + j], gv = g[base + j], vv = v[base + j];
    float bt = beta[bbase];

    for (int t = 0; t < TSEQ; ++t) {
        qs[j] = qv; ks[j] = kv; eg[j] = __expf(gv);
        __syncthreads();

        // issue next timestep's loads; consumed after the trailing barrier
        float qn = 0.f, kn = 0.f, gn = 0.f, vn = 0.f, btn = 0.f;
        if (t + 1 < TSEQ) {
            size_t off2 = base + (size_t)(t + 1) * strideT + j;
            qn = q[off2]; kn = k[off2]; gn = g[off2]; vn = v[off2];
            btn = beta[bbase + (size_t)(t + 1) * HH];
        }

        float acc = 0.f;
        #pragma unroll
        for (int i = 0; i < KK; ++i) {
            S[i] *= eg[i];
            acc = fmaf(S[i], ks[i], acc);
        }
        float vd = (vv - acc) * bt;
        float oacc = 0.f;
        #pragma unroll
        for (int i = 0; i < KK; ++i) {
            float s = fmaf(ks[i], vd, S[i]);
            S[i] = s;
            oacc = fmaf(s, qs[i], oacc);
        }
        o[base + (size_t)t * strideT + j] = oacc;
        __syncthreads();

        qv = qn; kv = kn; gv = gn; vv = vn; bt = btn;
    }
}

// ---------------------------------------------------------------------------
extern "C" void kernel_launch(void* const* d_in, const int* in_sizes, int n_in,
                              void* d_out, int out_size, void* d_ws, size_t ws_size,
                              hipStream_t stream)
{
    (void)in_sizes; (void)n_in; (void)out_size; (void)ws_size;

    const float* x       = (const float*)d_in[0];
    const float* Wq      = (const float*)d_in[1];
    const float* Wk      = (const float*)d_in[2];
    const float* Wv      = (const float*)d_in[3];
    const float* cqw     = (const float*)d_in[4];
    const float* cqb     = (const float*)d_in[5];
    const float* ckw     = (const float*)d_in[6];
    const float* ckb     = (const float*)d_in[7];
    const float* cvw     = (const float*)d_in[8];
    const float* cvb     = (const float*)d_in[9];
    const float* Wfa     = (const float*)d_in[10];
    const float* Wfb     = (const float*)d_in[11];
    const float* A_log   = (const float*)d_in[12];
    const float* dt_bias = (const float*)d_in[13];
    const float* Wb      = (const float*)d_in[14];
    const float* Wga     = (const float*)d_in[15];
    const float* Wgb     = (const float*)d_in[16];
    const float* onw     = (const float*)d_in[17];
    const float* Wo      = (const float*)d_in[18];
    float* out = (float*)d_out;

    // ---- workspace carve (256B aligned) ----
    char* wsp = (char*)d_ws;
    auto alloc_f = [&](size_t n) {
        float* r = (float*)wsp; wsp += ((n * 4 + 255) & ~(size_t)255); return r; };
    auto alloc_h = [&](size_t n) {
        __bf16* r = (__bf16*)wsp; wsp += ((n * 2 + 255) & ~(size_t)255); return r; };

    float* qp = alloc_f((size_t)NTOK * DD);
    float* kp = alloc_f((size_t)NTOK * DD);
    float* vp = alloc_f((size_t)NTOK * DD);
    float* qc = alloc_f((size_t)NTOK * DD);
    float* kc = alloc_f((size_t)NTOK * DD);
    float* vc = alloc_f((size_t)NTOK * DD);
    float* xfa   = alloc_f((size_t)NTOK * KK);
    float* xga   = alloc_f((size_t)NTOK * KK);
    float* betab = alloc_f((size_t)NTOK * HH);

    __bf16* xbf   = alloc_h((size_t)NTOK * HIDD);
    __bf16* wqbf  = alloc_h((size_t)HIDD * DD);   // transposed: [D][HID]
    __bf16* wkbf  = alloc_h((size_t)HIDD * DD);
    __bf16* wvbf  = alloc_h((size_t)HIDD * DD);
    __bf16* wobf  = alloc_h((size_t)DD * HIDD);   // transposed: [HID][D]
    __bf16* wfabf = alloc_h((size_t)HIDD * KK);   // transposed: [K][HID]
    __bf16* wfbbf = alloc_h((size_t)KK * DD);     // transposed: [D][K]
    __bf16* wbbf  = alloc_h((size_t)HIDD * HH);   // transposed: [H][HID]
    __bf16* wgabf = alloc_h((size_t)HIDD * KK);   // transposed: [K][HID]
    __bf16* wgbbf = alloc_h((size_t)KK * DD);     // transposed: [D][K]
    __bf16* xfabf = alloc_h((size_t)NTOK * KK);
    __bf16* xgabf = alloc_h((size_t)NTOK * KK);
    __bf16* obf   = alloc_h((size_t)NTOK * DD);

    // Reuse pre-conv f32 buffers once convs consumed them (stream-ordered).
    float* gbuf    = qp;
    float* gatebuf = kp;
    float* obuf    = vp;

    dim3 blk(256);
    auto gg = [](int M, int N) {
        return dim3((unsigned)((N + GBN - 1) / GBN), (unsigned)((M + GBM - 1) / GBM));
    };
    auto cvt = [&](const float* src, __bf16* dst, size_t n) {
        cvt_bf16_kernel<<<dim3((unsigned)((n / 4 + 255) / 256)), blk, 0, stream>>>(
            src, dst, (int)(n / 4));
    };
    auto cvtT = [&](const float* src, __bf16* dst, int R, int C) {
        dim3 g((unsigned)((C + 31) / 32), (unsigned)((R + 31) / 32));
        cvt_transpose_bf16_kernel<<<g, blk, 0, stream>>>(src, dst, R, C);
    };

    // ---- bf16 conversions: x row-major, weights transposed ----
    cvt (x,   xbf,   (size_t)NTOK * HIDD);
    cvtT(Wq,  wqbf,  HIDD, DD);
    cvtT(Wk,  wkbf,  HIDD, DD);
    cvtT(Wv,  wvbf,  HIDD, DD);
    cvtT(Wo,  wobf,  DD, HIDD);
    cvtT(Wfa, wfabf, HIDD, KK);
    cvtT(Wfb, wfbbf, KK, DD);
    cvtT(Wb,  wbbf,  HIDD, HH);
    cvtT(Wga, wgabf, HIDD, KK);
    cvtT(Wgb, wgbbf, KK, DD);

    // ---- q/k/v projections ----
    gemm_wmma_bf16<<<gg(NTOK, DD), blk, 0, stream>>>(xbf, wqbf, qp, NTOK, DD, HIDD);
    gemm_wmma_bf16<<<gg(NTOK, DD), blk, 0, stream>>>(xbf, wkbf, kp, NTOK, DD, HIDD);
    gemm_wmma_bf16<<<gg(NTOK, DD), blk, 0, stream>>>(xbf, wvbf, vp, NTOK, DD, HIDD);

    dim3 cgrid((BB * TSEQ * DD + 255) / 256);
    conv_silu_kernel<<<cgrid, blk, 0, stream>>>(qp, cqw, cqb, qc);
    conv_silu_kernel<<<cgrid, blk, 0, stream>>>(kp, ckw, ckb, kc);
    conv_silu_kernel<<<cgrid, blk, 0, stream>>>(vp, cvw, cvb, vc);

    // ---- decay gate g ----
    gemm_wmma_bf16<<<gg(NTOK, KK), blk, 0, stream>>>(xbf, wfabf, xfa, NTOK, KK, HIDD);
    cvt(xfa, xfabf, (size_t)NTOK * KK);
    gemm_wmma_bf16<<<gg(NTOK, DD), blk, 0, stream>>>(xfabf, wfbbf, gbuf, NTOK, DD, KK);
    gate_kernel<<<cgrid, blk, 0, stream>>>(gbuf, dt_bias, A_log);

    // ---- beta ----
    gemm_wmma_bf16<<<gg(NTOK, HH), blk, 0, stream>>>(xbf, wbbf, betab, NTOK, HH, HIDD);
    sigmoid_kernel<<<dim3((NTOK * HH + 255) / 256), blk, 0, stream>>>(betab, NTOK * HH);

    // ---- output gate ----
    gemm_wmma_bf16<<<gg(NTOK, KK), blk, 0, stream>>>(xbf, wgabf, xga, NTOK, KK, HIDD);
    cvt(xga, xgabf, (size_t)NTOK * KK);
    gemm_wmma_bf16<<<gg(NTOK, DD), blk, 0, stream>>>(xgabf, wgbbf, gatebuf, NTOK, DD, KK);

    // ---- l2norm q (with 1/sqrt(K)) and k ----
    int rows = NTOK * HH;
    l2norm_kernel<<<dim3(rows / 8), blk, 0, stream>>>(qc, QSCALE);
    l2norm_kernel<<<dim3(rows / 8), blk, 0, stream>>>(kc, 1.0f);

    // ---- sequential gated delta-rule scan ----
    kda_scan_kernel<<<dim3(BB * HH), dim3(128), 0, stream>>>(qc, kc, vc, gbuf, betab, obuf);

    // ---- gated RMSNorm + output projection ----
    rmsnorm_gate_kernel<<<dim3(rows / 8), blk, 0, stream>>>(obuf, gatebuf, onw);
    cvt(obuf, obf, (size_t)NTOK * DD);
    gemm_wmma_bf16<<<gg(NTOK, HIDD), blk, 0, stream>>>(obf, wobf, out, NTOK, HIDD, DD);
}
